// ConformerBlock_76493367542435
// MI455X (gfx1250) — compile-verified, matrix operand order checked by reading, not run on hardware
//
#include <hip/hip_runtime.h>
#include <cstdint>

#define DEV __device__ __forceinline__
typedef unsigned short u16;

typedef __attribute__((ext_vector_type(16))) __bf16 bf16x16;
typedef __attribute__((ext_vector_type(8)))  float  v8f;
typedef unsigned int v4u __attribute__((ext_vector_type(4)));
typedef int v8i __attribute__((ext_vector_type(8)));
typedef int v4i __attribute__((ext_vector_type(4)));

union FragU { bf16x16 v; uint4 q[2]; };

DEV v8f v8zero() { v8f z = {0.f,0.f,0.f,0.f,0.f,0.f,0.f,0.f}; return z; }

DEV bf16x16 frag_from(const void* p0, const void* p1) {
  FragU f; f.q[0] = *(const uint4*)p0; f.q[1] = *(const uint4*)p1; return f.v;
}

// A/B fragment loader for 16x16x32 bf16 WMMA from a row-major (K-contiguous) tile.
DEV bf16x16 lds_frag(const u16* tile, int rowbase, int kk, int stride, int lane) {
  int r = rowbase + (lane & 15);
  int o = kk + ((lane & 16) ? 8 : 0);
  const u16* p = tile + (size_t)r * stride + o;
  return frag_from(p, p + 16);
}

DEV v8f wmma_bf16(bf16x16 a, bf16x16 b, v8f c) {
  return __builtin_amdgcn_wmma_f32_16x16x32_bf16(false, a, false, b, (short)0, c, false, false);
}

DEV u16 f2bf(float x) {
  unsigned int u = __builtin_bit_cast(unsigned int, x);
  u += 0x7fffu + ((u >> 16) & 1u);
  return (u16)(u >> 16);
}
DEV float bf2f(u16 h) {
  unsigned int u = ((unsigned int)h) << 16;
  return __builtin_bit_cast(float, u);
}
DEV float sigm(float x) { return 1.f / (1.f + __expf(-x)); }

// TDM: async 2-D tile load (64 K-elems wide, `rows` tall, bf16) global -> LDS,
// with HW padding of 4 DWORDs every 32 DWORDs => padded LDS row stride of 72 u16.
DEV void tdm_load_tile(unsigned lds_off, const u16* gptr, unsigned Ktot,
                       unsigned rows, unsigned tensor_rows) {
  unsigned long long ga = (unsigned long long)(uintptr_t)gptr;
  v4u g0;
  g0[0] = 1u;                                               // count = 1 valid descriptor
  g0[1] = lds_off;                                          // LDS byte address
  g0[2] = (unsigned)ga;                                     // global_addr[31:0]
  g0[3] = (unsigned)((ga >> 32) & 0x1ffffffu) | (2u << 30); // global_addr[56:32] | type=2
  v8i g1;
  g1[0] = (1 << 16) | (1 << 20) | (4 << 22) | (3 << 25);    // 2B elems, pad_en, 32DW interval, 4DW pad
  g1[1] = (int)((Ktot & 0xffffu) << 16);                    // tensor_dim0[15:0]
  g1[2] = (int)(((Ktot >> 16) & 0xffffu) | ((tensor_rows & 0xffffu) << 16)); // dim0[31:16]|dim1[15:0]
  g1[3] = (int)(((tensor_rows >> 16) & 0xffffu) | (64u << 16));              // dim1[31:16]|tile_dim0=64
  g1[4] = (int)(rows & 0xffffu);                            // tile_dim1 (tile_dim2 = 0 -> 2D)
  g1[5] = (int)Ktot;                                        // tensor_dim0_stride[31:0]
  g1[6] = 0;                                                // stride0[47:32] | stride1[15:0]
  g1[7] = 0;
  v4i z4 = {0, 0, 0, 0};
#if __clang_major__ >= 23
  v8i z8 = {0, 0, 0, 0, 0, 0, 0, 0};
  __builtin_amdgcn_tensor_load_to_lds(g0, g1, z4, z4, z8, 0);
#else
  __builtin_amdgcn_tensor_load_to_lds(g0, g1, z4, z4, 0);
#endif
}

// ---------------------------------------------------------------- constants
static constexpr int Bz = 16, Nn = 512, Dd = 512, Hh = 8, DH = 64;
static constexpr int FF = 2048, CI = 1024, MT = Bz * Nn;   // 8192 rows
enum { EP_STORE_BF16 = 0, EP_BIAS_SWISH_BF16, EP_BIAS_BF16,
       EP_BIAS_HALFRES_F32, EP_RES_F32, EP_BIAS_RES_F32 };

// ---------------------------------------------------------------- GEMM
// C[M,N] = A[M,K] * Bt[N,K]^T, bf16 in / f32 accum, TDM double-buffered staging.
template<int EPI>
__global__ __launch_bounds__(256) void gemm_bf16(
    const u16* __restrict__ A, const u16* __restrict__ Bt,
    const float* __restrict__ bias, const float* __restrict__ resid,
    u16* __restrict__ outb, float* __restrict__ outf,
    int Ntot, int Ktot)
{
  __shared__ __align__(16) u16 la[2][128 * 72];
  __shared__ __align__(16) u16 lb[2][64 * 72];
  const int tid = threadIdx.x, lane = tid & 31, wid = tid >> 5;
  const int mw = wid >> 1, nw = wid & 1;
  const int m0 = blockIdx.y * 128, n0 = blockIdx.x * 64;
  const int nk = Ktot >> 6;

  if (wid == 0) {   // TDM issues once per wave (EXEC-independent); wave 0 drives DMA
    tdm_load_tile((unsigned)(uintptr_t)(void*)la[0], A + (size_t)m0 * Ktot, Ktot, 128, MT);
    tdm_load_tile((unsigned)(uintptr_t)(void*)lb[0], Bt + (size_t)n0 * Ktot, Ktot, 64, Ntot);
  }

  v8f acc[2][2];
  acc[0][0] = v8zero(); acc[0][1] = v8zero(); acc[1][0] = v8zero(); acc[1][1] = v8zero();

  for (int kb = 0; kb < nk; ++kb) {
    const int cur = kb & 1;
    if (wid == 0) __builtin_amdgcn_s_wait_tensorcnt(0);
    __syncthreads();                        // tiles[cur] visible to all waves
    if (wid == 0 && kb + 1 < nk) {          // prefetch next tiles into the other buffer
      tdm_load_tile((unsigned)(uintptr_t)(void*)la[cur ^ 1],
                    A + (size_t)m0 * Ktot + (kb + 1) * 64, Ktot, 128, MT);
      tdm_load_tile((unsigned)(uintptr_t)(void*)lb[cur ^ 1],
                    Bt + (size_t)n0 * Ktot + (kb + 1) * 64, Ktot, 64, Ntot);
    }
#pragma unroll
    for (int ks = 0; ks < 2; ++ks) {
      bf16x16 a0 = lds_frag(la[cur], mw * 32,      ks * 32, 72, lane);
      bf16x16 a1 = lds_frag(la[cur], mw * 32 + 16, ks * 32, 72, lane);
      bf16x16 b0 = lds_frag(lb[cur], nw * 32,      ks * 32, 72, lane);
      bf16x16 b1 = lds_frag(lb[cur], nw * 32 + 16, ks * 32, 72, lane);
      acc[0][0] = wmma_bf16(a0, b0, acc[0][0]);
      acc[0][1] = wmma_bf16(a0, b1, acc[0][1]);
      acc[1][0] = wmma_bf16(a1, b0, acc[1][0]);
      acc[1][1] = wmma_bf16(a1, b1, acc[1][1]);
    }
  }

  constexpr bool HAS_BIAS = (EPI == EP_BIAS_SWISH_BF16 || EPI == EP_BIAS_BF16 ||
                             EPI == EP_BIAS_HALFRES_F32 || EPI == EP_BIAS_RES_F32);
  const int grp = (lane >> 4) & 1, cl = lane & 15;
#pragma unroll
  for (int ti = 0; ti < 2; ++ti)
#pragma unroll
    for (int tj = 0; tj < 2; ++tj) {
      int gm = m0 + mw * 32 + ti * 16;
      int col = n0 + nw * 32 + tj * 16 + cl;
      float bv = HAS_BIAS ? bias[col] : 0.f;
#pragma unroll
      for (int r = 0; r < 8; ++r) {
        int row = gm + r + grp * 8;
        size_t idx = (size_t)row * Ntot + col;
        float v = acc[ti][tj][r] + bv;
        if (EPI == EP_STORE_BF16)            outb[idx] = f2bf(v);
        else if (EPI == EP_BIAS_SWISH_BF16)  outb[idx] = f2bf(v * sigm(v));
        else if (EPI == EP_BIAS_BF16)        outb[idx] = f2bf(v);
        else if (EPI == EP_BIAS_HALFRES_F32) outf[idx] = 0.5f * v + resid[idx];
        else if (EPI == EP_RES_F32)          outf[idx] = v + resid[idx];
        else if (EPI == EP_BIAS_RES_F32)     outf[idx] = v + resid[idx];
      }
    }
}

// ---------------------------------------------------------------- LayerNorm (1 wave / 512-row)
template<bool BF16OUT>
__global__ __launch_bounds__(256) void ln_kernel(
    const float* __restrict__ x, const float* __restrict__ g, const float* __restrict__ bt,
    u16* __restrict__ outb, float* __restrict__ outf)
{
  const int lane = threadIdx.x & 31;
  const size_t row = (size_t)blockIdx.x * 8 + (threadIdx.x >> 5);
  const float* xr = x + row * 512;
  float4 v[4]; float s = 0.f, s2 = 0.f;
#pragma unroll
  for (int j = 0; j < 4; ++j) {
    v[j] = *(const float4*)(xr + j * 128 + lane * 4);
    s  += v[j].x + v[j].y + v[j].z + v[j].w;
    s2 += v[j].x*v[j].x + v[j].y*v[j].y + v[j].z*v[j].z + v[j].w*v[j].w;
  }
#pragma unroll
  for (int m = 1; m < 32; m <<= 1) { s += __shfl_xor(s, m, 32); s2 += __shfl_xor(s2, m, 32); }
  float mean = s * (1.f / 512.f);
  float rstd = rsqrtf(s2 * (1.f / 512.f) - mean * mean + 1e-5f);
#pragma unroll
  for (int j = 0; j < 4; ++j) {
    int col = j * 128 + lane * 4;
    float4 gg = *(const float4*)(g + col);
    float4 bb = *(const float4*)(bt + col);
    float y0 = (v[j].x - mean) * rstd * gg.x + bb.x;
    float y1 = (v[j].y - mean) * rstd * gg.y + bb.y;
    float y2 = (v[j].z - mean) * rstd * gg.z + bb.z;
    float y3 = (v[j].w - mean) * rstd * gg.w + bb.w;
    if (BF16OUT) {
      u16* o = outb + row * 512 + col;
      o[0] = f2bf(y0); o[1] = f2bf(y1); o[2] = f2bf(y2); o[3] = f2bf(y3);
    } else {
      float4 o4 = {y0, y1, y2, y3};
      *(float4*)(outf + row * 512 + col) = o4;
    }
  }
}

// ---------------------------------------------------------------- attention
__global__ __launch_bounds__(128) void attn_kernel(
    const u16* __restrict__ qkv,   // [8192][1536] bf16: q|k|v per head-major cols
    const u16* __restrict__ erel,  // [1025][64] bf16
    u16* __restrict__ outab)       // [8192][512] bf16
{
  __shared__ __align__(16) u16 q_lds[64 * 72];
  __shared__ __align__(16) u16 k_lds[32 * 72];
  __shared__ __align__(16) u16 vt_lds[64 * 40];     // V transposed: [d][m]
  __shared__ __align__(16) float qe_lds[4][16 * 64];
  __shared__ __align__(16) u16 p_lds[4][16 * 40];

  const int tid = threadIdx.x, lane = tid & 31, wid = tid >> 5;
  const int blk = blockIdx.x;
  const int nb = blk & 7, h = (blk >> 3) & 7, b = blk >> 6;
  const int n0wg = nb * 64, nq0 = n0wg + wid * 16;
  const int grp = (lane >> 4) & 1, cl = lane & 15;
  const float scale = 0.125f;

#pragma unroll
  for (int i = 0; i < 4; ++i) {
    int lin = tid + i * 128, row = lin >> 3, cq = lin & 7;
    *(uint4*)(q_lds + row * 72 + cq * 8) =
        *(const uint4*)(qkv + (size_t)(b * 512 + n0wg + row) * 1536 + h * 64 + cq * 8);
  }
  __syncthreads();
  bf16x16 aq0 = lds_frag(q_lds, wid * 16, 0,  72, lane);
  bf16x16 aq1 = lds_frag(q_lds, wid * 16, 32, 72, lane);

  float rm[8], rl[8];
  v8f o[4];
#pragma unroll
  for (int r = 0; r < 8; ++r) { rm[r] = -1e30f; rl[r] = 0.f; }
  o[0] = v8zero(); o[1] = v8zero(); o[2] = v8zero(); o[3] = v8zero();

  for (int c = 0; c < 16; ++c) {
    const int m0 = c * 32;
    __syncthreads();
#pragma unroll
    for (int i = 0; i < 2; ++i) {   // K chunk, row-major (serves directly as B^T)
      int lin = tid + i * 128, row = lin >> 3, cq = lin & 7;
      *(uint4*)(k_lds + row * 72 + cq * 8) =
          *(const uint4*)(qkv + (size_t)(b * 512 + m0 + row) * 1536 + 512 + h * 64 + cq * 8);
    }
#pragma unroll
    for (int i = 0; i < 16; ++i) {  // V chunk transposed -> [d][m]
      int lin = tid + i * 128, m = lin >> 6, d = lin & 63;
      vt_lds[d * 40 + m] = qkv[(size_t)(b * 512 + m0 + m) * 1536 + 1024 + h * 64 + d];
    }
    __syncthreads();

    // rel-pos window: qe[i][u] = q[nq0+i] . E[t0+u],  t0 = nq0-m0+481, u in [0,64)
    const int t0 = nq0 - m0 + 481;
    v8f qe[4];
    qe[0] = v8zero(); qe[1] = v8zero(); qe[2] = v8zero(); qe[3] = v8zero();
#pragma unroll
    for (int nt = 0; nt < 4; ++nt) {
      int tt = t0 + nt * 16 + cl;
      if (tt < 0) tt = 0; if (tt > 1024) tt = 1024;
      const u16* ep0 = erel + (size_t)tt * 64 + ((lane & 16) ? 8 : 0);
      qe[nt] = wmma_bf16(aq0, frag_from(ep0,      ep0 + 16), qe[nt]);
      qe[nt] = wmma_bf16(aq1, frag_from(ep0 + 32, ep0 + 48), qe[nt]);
    }
    float* qp = qe_lds[wid];
#pragma unroll
    for (int nt = 0; nt < 4; ++nt)
#pragma unroll
      for (int r = 0; r < 8; ++r)
        qp[(r + grp * 8) * 64 + nt * 16 + cl] = qe[nt][r];

    // S = scale * (Q K^T + pos)
    v8f st[2];
#pragma unroll
    for (int tj = 0; tj < 2; ++tj) {
      v8f s = v8zero();
      s = wmma_bf16(aq0, lds_frag(k_lds, tj * 16, 0,  72, lane), s);
      s = wmma_bf16(aq1, lds_frag(k_lds, tj * 16, 32, 72, lane), s);
#pragma unroll
      for (int r = 0; r < 8; ++r) {
        int i = r + grp * 8, j = tj * 16 + cl;
        s[r] = scale * (s[r] + qp[i * 64 + (i - j + 31)]);
      }
      st[tj] = s;
    }

    // online softmax (rows live across 16-lane groups)
    float cm[8];
#pragma unroll
    for (int r = 0; r < 8; ++r) cm[r] = fmaxf(st[0][r], st[1][r]);
#pragma unroll
    for (int m = 1; m < 16; m <<= 1)
#pragma unroll
      for (int r = 0; r < 8; ++r) cm[r] = fmaxf(cm[r], __shfl_xor(cm[r], m, 32));
#pragma unroll
    for (int r = 0; r < 8; ++r) {
      float mn = fmaxf(rm[r], cm[r]);
      float al = __expf(rm[r] - mn);
      rm[r] = mn; rl[r] *= al;
      o[0][r] *= al; o[1][r] *= al; o[2][r] *= al; o[3][r] *= al;
    }
#pragma unroll
    for (int tj = 0; tj < 2; ++tj)
#pragma unroll
      for (int r = 0; r < 8; ++r) st[tj][r] = __expf(st[tj][r] - rm[r]);
#pragma unroll
    for (int r = 0; r < 8; ++r) {
      float su = st[0][r] + st[1][r];
#pragma unroll
      for (int m = 1; m < 16; m <<= 1) su += __shfl_xor(su, m, 32);
      rl[r] += su;
    }

    // P -> LDS (re-fragment C-layout to A-layout) then P x V
    u16* pp = p_lds[wid];
#pragma unroll
    for (int tj = 0; tj < 2; ++tj)
#pragma unroll
      for (int r = 0; r < 8; ++r)
        pp[(r + grp * 8) * 40 + tj * 16 + cl] = f2bf(st[tj][r]);
    bf16x16 ap = lds_frag(pp, 0, 0, 40, lane);
#pragma unroll
    for (int dt = 0; dt < 4; ++dt)
      o[dt] = wmma_bf16(ap, lds_frag(vt_lds, dt * 16, 0, 40, lane), o[dt]);
  }

#pragma unroll
  for (int dt = 0; dt < 4; ++dt)
#pragma unroll
    for (int r = 0; r < 8; ++r) {
      int i = r + grp * 8;
      size_t idx = (size_t)(b * 512 + nq0 + i) * 512 + h * 64 + dt * 16 + cl;
      outab[idx] = f2bf(o[dt][r] / rl[r]);
    }
}

// ---------------------------------------------------------------- GLU, depthwise conv, converts
__global__ void glu_kernel(const u16* __restrict__ p, u16* __restrict__ u) {
  const int total = MT * CI;
  for (int i = blockIdx.x * blockDim.x + threadIdx.x; i < total; i += gridDim.x * blockDim.x) {
    int m = i >> 10, o = i & 1023;
    float a = bf2f(p[(size_t)m * 2048 + o]);
    float g = bf2f(p[(size_t)m * 2048 + o + 1024]);
    u[i] = f2bf(a * sigm(g));
  }
}

__global__ __launch_bounds__(256) void dwconv_kernel(
    const u16* __restrict__ u, const float* __restrict__ w, const float* __restrict__ wb,
    const float* __restrict__ bg, const float* __restrict__ bb,
    const float* __restrict__ bm, const float* __restrict__ bv, u16* __restrict__ out)
{
  const int c = blockIdx.y * 256 + threadIdx.x;
  const int bn = blockIdx.x, n = bn & 511, rowbase = bn - n;
  float acc = 0.f;
#pragma unroll
  for (int k = 0; k < 31; ++k) {
    int nn = n + k - 15;
    if (nn >= 0 && nn < 512)
      acc += bf2f(u[(size_t)(rowbase + nn) * 1024 + c]) * w[c * 31 + k];
  }
  acc += wb[c];
  acc = (acc - bm[c]) * rsqrtf(bv[c] + 1e-5f) * bg[c] + bb[c];
  acc = acc * sigm(acc);
  out[(size_t)bn * 1024 + c] = f2bf(acc);
}

__global__ void convT_kernel(const float* __restrict__ in, u16* __restrict__ out, int R, int C) {
  const int total = R * C;
  for (int i = blockIdx.x * blockDim.x + threadIdx.x; i < total; i += gridDim.x * blockDim.x) {
    int c = i / R, r = i - c * R;
    out[i] = f2bf(in[(size_t)r * C + c]);
  }
}
__global__ void convC_kernel(const float* __restrict__ in, u16* __restrict__ out, int n) {
  for (int i = blockIdx.x * blockDim.x + threadIdx.x; i < n; i += gridDim.x * blockDim.x)
    out[i] = f2bf(in[i]);
}

// ---------------------------------------------------------------- launcher
extern "C" void kernel_launch(void* const* d_in, const int* in_sizes, int n_in,
                              void* d_out, int out_size, void* d_ws, size_t ws_size,
                              hipStream_t stream) {
  const float* xin      = (const float*)d_in[0];
  const float* ff1_g    = (const float*)d_in[1];  const float* ff1_b   = (const float*)d_in[2];
  const float* ff1_w1   = (const float*)d_in[3];  const float* ff1_b1  = (const float*)d_in[4];
  const float* ff1_w2   = (const float*)d_in[5];  const float* ff1_b2  = (const float*)d_in[6];
  const float* at_g     = (const float*)d_in[7];  const float* at_b    = (const float*)d_in[8];
  const float* qkv_w    = (const float*)d_in[9];  const float* out_w   = (const float*)d_in[10];
  const float* rel_emb  = (const float*)d_in[11];
  const float* cv_g     = (const float*)d_in[12]; const float* cv_b    = (const float*)d_in[13];
  const float* pw1_w    = (const float*)d_in[14]; const float* pw1_b   = (const float*)d_in[15];
  const float* dw_w     = (const float*)d_in[16]; const float* dw_b    = (const float*)d_in[17];
  const float* bn_g     = (const float*)d_in[18]; const float* bn_b    = (const float*)d_in[19];
  const float* bn_m     = (const float*)d_in[20]; const float* bn_v    = (const float*)d_in[21];
  const float* pw2_w    = (const float*)d_in[22]; const float* pw2_b   = (const float*)d_in[23];
  const float* ff2_g    = (const float*)d_in[24]; const float* ff2_b   = (const float*)d_in[25];
  const float* ff2_w1   = (const float*)d_in[26]; const float* ff2_b1  = (const float*)d_in[27];
  const float* ff2_w2   = (const float*)d_in[28]; const float* ff2_b2  = (const float*)d_in[29];
  const float* po_g     = (const float*)d_in[30]; const float* po_b    = (const float*)d_in[31];

  char* p = (char*)d_ws;
  auto alloc = [&](size_t bytes) { void* r = p; p += (bytes + 255) & ~(size_t)255; return r; };
  float* x_cur = (float*)alloc((size_t)MT * 512 * 4);
  u16* hb   = (u16*)alloc((size_t)MT * 512 * 2);
  u16* big  = (u16*)alloc((size_t)MT * 2048 * 2);
  u16* ub   = (u16*)alloc((size_t)MT * 1024 * 2);
  u16* wbuf = (u16*)alloc((size_t)MT * 1024 * 2);
  u16* wt11 = (u16*)alloc(512 * 2048 * 2);
  u16* wt12 = (u16*)alloc(512 * 2048 * 2);
  u16* wtqk = (u16*)alloc(512 * 1536 * 2);
  u16* wtot = (u16*)alloc(512 * 512 * 2);
  u16* wtp1 = (u16*)alloc(2048 * 512 * 2);
  u16* wtp2 = (u16*)alloc(512 * 1024 * 2);
  u16* wt21 = (u16*)alloc(512 * 2048 * 2);
  u16* wt22 = (u16*)alloc(512 * 2048 * 2);
  u16* ebf  = (u16*)alloc(1025 * 64 * 2);

  hipMemcpyAsync(x_cur, xin, (size_t)MT * 512 * 4, hipMemcpyDeviceToDevice, stream);

  // weight casts / transposes to B^T (K-contiguous) bf16
  convT_kernel<<<4096, 256, 0, stream>>>(ff1_w1, wt11, 512, 2048);
  convT_kernel<<<4096, 256, 0, stream>>>(ff1_w2, wt12, 2048, 512);
  convT_kernel<<<3072, 256, 0, stream>>>(qkv_w,  wtqk, 512, 1536);
  convT_kernel<<<1024, 256, 0, stream>>>(out_w,  wtot, 512, 512);
  convC_kernel<<<4096, 256, 0, stream>>>(pw1_w,  wtp1, 2048 * 512);
  convC_kernel<<<2048, 256, 0, stream>>>(pw2_w,  wtp2, 512 * 1024);
  convT_kernel<<<4096, 256, 0, stream>>>(ff2_w1, wt21, 512, 2048);
  convT_kernel<<<4096, 256, 0, stream>>>(ff2_w2, wt22, 2048, 512);
  convC_kernel<<<257, 256, 0, stream>>>(rel_emb, ebf, 1025 * 64);

  // FF1 (x = 0.5*ff(x) + x)
  ln_kernel<true><<<1024, 256, 0, stream>>>(x_cur, ff1_g, ff1_b, hb, nullptr);
  gemm_bf16<EP_BIAS_SWISH_BF16><<<dim3(32, 64), 256, 0, stream>>>(hb, wt11, ff1_b1, nullptr, big, nullptr, 2048, 512);
  gemm_bf16<EP_BIAS_HALFRES_F32><<<dim3(8, 64), 256, 0, stream>>>(big, wt12, ff1_b2, x_cur, nullptr, x_cur, 512, 2048);

  // attention
  ln_kernel<true><<<1024, 256, 0, stream>>>(x_cur, at_g, at_b, hb, nullptr);
  gemm_bf16<EP_STORE_BF16><<<dim3(24, 64), 256, 0, stream>>>(hb, wtqk, nullptr, nullptr, big, nullptr, 1536, 512);
  attn_kernel<<<1024, 128, 0, stream>>>(big, ebf, ub);
  gemm_bf16<EP_RES_F32><<<dim3(8, 64), 256, 0, stream>>>(ub, wtot, nullptr, x_cur, nullptr, x_cur, 512, 512);

  // conv module
  ln_kernel<true><<<1024, 256, 0, stream>>>(x_cur, cv_g, cv_b, hb, nullptr);
  gemm_bf16<EP_BIAS_BF16><<<dim3(32, 64), 256, 0, stream>>>(hb, wtp1, pw1_b, nullptr, big, nullptr, 2048, 512);
  glu_kernel<<<4096, 256, 0, stream>>>(big, ub);
  dwconv_kernel<<<dim3(8192, 4), 256, 0, stream>>>(ub, dw_w, dw_b, bn_g, bn_b, bn_m, bn_v, wbuf);
  gemm_bf16<EP_BIAS_RES_F32><<<dim3(8, 64), 256, 0, stream>>>(wbuf, wtp2, pw2_b, x_cur, nullptr, x_cur, 512, 1024);

  // FF2
  ln_kernel<true><<<1024, 256, 0, stream>>>(x_cur, ff2_g, ff2_b, hb, nullptr);
  gemm_bf16<EP_BIAS_SWISH_BF16><<<dim3(32, 64), 256, 0, stream>>>(hb, wt21, ff2_b1, nullptr, big, nullptr, 2048, 512);
  gemm_bf16<EP_BIAS_HALFRES_F32><<<dim3(8, 64), 256, 0, stream>>>(big, wt22, ff2_b2, x_cur, nullptr, x_cur, 512, 2048);

  // post-norm -> d_out (f32)
  ln_kernel<false><<<1024, 256, 0, stream>>>(x_cur, po_g, po_b, nullptr, (float*)d_out);

  (void)in_sizes; (void)n_in; (void)out_size; (void)ws_size;
}
